// VQ_VAE_78426102825472
// MI455X (gfx1250) — matrix operand order repaired and checked by reading, not compiled, not run on hardware
//
#include <hip/hip_runtime.h>
#include <hip/hip_bf16.h>

typedef __attribute__((ext_vector_type(16))) _Float16 v16h;
typedef __attribute__((ext_vector_type(8)))  float    v8f;
typedef __attribute__((ext_vector_type(4)))  unsigned u32x4;
typedef __attribute__((ext_vector_type(8)))  unsigned u32x8;

// ---------------- problem constants ----------------
static constexpr int BATCH  = 256;
static constexpr int KCODES = 8192;       // codebook entries
static constexpr int EDIM   = 64;         // code dim
static constexpr int NROWS  = BATCH * 32; // B*L = 8192 pred rows

// =====================================================================
// Tensor Data Mover: 2-D tile load global->LDS.
// D# group0 (4 SGPRs): count=1 | lds_addr | global_addr[56:0] | type=2
// D# group1 (8 SGPRs): data_size=4B, tensor_dim0/1, tile_dim0/1, dim0_stride.
// VADDR2/VADDR3 omitted (NULL) -> 2D tensor form.
// =====================================================================
static __device__ __forceinline__ void tdm_load_tile_2d(
        unsigned ldsOff, const void* gaddr,
        unsigned tensor_d0, unsigned tensor_d1,
        unsigned tile_d0, unsigned tile_d1,
        unsigned long long stride0_elems) {
    const unsigned long long ga = (unsigned long long)gaddr;
    u32x4 g0;
    g0[0] = 1u;                                        // count=1 (valid), user mode
    g0[1] = ldsOff;                                    // lds_addr (bytes)
    g0[2] = (unsigned)(ga & 0xffffffffu);              // global_addr[31:0]
    g0[3] = (unsigned)((ga >> 32) & 0x01ffffffu)       // global_addr[56:32]
            | (2u << 30);                              // type=2 ("image")
    u32x8 g1;
    g1[0] = 2u << 16;                                  // data_size: 2 -> 4 bytes
    g1[1] = (tensor_d0 & 0xffffu) << 16;               // tensor_dim0[15:0]
    g1[2] = (tensor_d0 >> 16) | ((tensor_d1 & 0xffffu) << 16);
    g1[3] = (tensor_d1 >> 16) | (tile_d0 << 16);       // tile_dim0
    g1[4] = tile_d1 & 0xffffu;                         // tile_dim1 (tile_dim2=0)
    g1[5] = (unsigned)(stride0_elems & 0xffffffffu);   // tensor_dim0_stride[31:0]
    g1[6] = (unsigned)((stride0_elems >> 32) & 0xffffu); // stride[47:32]
    g1[7] = 0u;                                        // tensor_dim1_stride unused (2D)
    asm volatile("tensor_load_to_lds %0, %1" :: "s"(g0), "s"(g1) : "memory");
}

// =====================================================================
// TDM-staged, double-buffered WMMA GEMM. Workgroup = 4 waves -> 64x64
// C-tile, each wave a 32x32 sub-tile (2x2 fragments). Wave0 keeps the
// next K-chunk's A/B tiles in flight on the Tensor Data Mover while all
// waves compute the current chunk from LDS (8 WMMAs per chunk per wave).
// TDM ops are in-order per wave, so waiting TENSORcnt<=2 after issuing
// the next pair guarantees the current pair has landed.
// =====================================================================
__global__ void __launch_bounds__(128)
wmma_gemm_tdm_kernel(const float* __restrict__ A,
                     const float* __restrict__ Bm,
                     const float* __restrict__ bias,
                     int biasMask,                 // period-1 (pow2)
                     float* __restrict__ C,
                     int M, int N, int Kd, int relu) {
    __shared__ float As[2][64 * 64];   // [buf][64 rows x 64 K]
    __shared__ float Bs[2][64 * 64];   // [buf][64 K x 64 cols]

    const int tid  = threadIdx.x;
    const int lane = tid & 31;
    const int wav  = tid >> 5;      // 0..3
    const int m16  = lane & 15;
    const int hi   = lane >> 4;
    const int wr   = (wav >> 1) * 32;   // wave row offset in 64x64 tile
    const int wc   = (wav & 1) * 32;    // wave col offset
    const int row0 = blockIdx.y * 64;
    const int col0 = blockIdx.x * 64;

    v8f acc00 = {}, acc01 = {}, acc10 = {}, acc11 = {};

    const int nch = Kd / 64;

    if (tid < 32) {   // prologue: chunk 0 into buffer 0
        tdm_load_tile_2d((unsigned)(unsigned long long)(const void*)&As[0][0],
                         A + (size_t)row0 * Kd, (unsigned)Kd, (unsigned)M,
                         64u, 64u, (unsigned long long)Kd);
        tdm_load_tile_2d((unsigned)(unsigned long long)(const void*)&Bs[0][0],
                         Bm + col0, (unsigned)N, (unsigned)Kd,
                         64u, 64u, (unsigned long long)N);
    }

    for (int i = 0; i < nch; ++i) {
        if (tid < 32) {
            if (i + 1 < nch) {      // keep next chunk in flight on the TDM
                const int kb = (i + 1) * 64;
                const int nb = (i + 1) & 1;
                tdm_load_tile_2d((unsigned)(unsigned long long)(const void*)&As[nb][0],
                                 A + (size_t)row0 * Kd + kb, (unsigned)Kd, (unsigned)M,
                                 64u, 64u, (unsigned long long)Kd);
                tdm_load_tile_2d((unsigned)(unsigned long long)(const void*)&Bs[nb][0],
                                 Bm + (size_t)kb * N + col0, (unsigned)N, (unsigned)Kd,
                                 64u, 64u, (unsigned long long)N);
                __builtin_amdgcn_s_wait_tensorcnt((short)2);  // current pair landed
            } else {
                __builtin_amdgcn_s_wait_tensorcnt((short)0);
            }
        }
        __syncthreads();            // current chunk visible to all waves

        const float* Abuf = &As[i & 1][0];
        const float* Bbuf = &Bs[i & 1][0];
#pragma unroll
        for (int kk = 0; kk < 64; kk += 32) {
            v16h a0, a1, b0, b1;
            const float* ap0 = Abuf + (size_t)(wr + m16) * 64 + kk + hi * 16;
            const float* ap1 = ap0 + 16 * 64;
#pragma unroll
            for (int j = 0; j < 8; ++j) {
                a0[j]     = (_Float16)ap0[j];       // K = kk + hi*16 + j
                a0[8 + j] = (_Float16)ap0[8 + j];   // K = kk + hi*16 + 8 + j
                a1[j]     = (_Float16)ap1[j];
                a1[8 + j] = (_Float16)ap1[8 + j];
            }
            const float* bp0 = Bbuf + (size_t)(kk + hi * 16) * 64 + (wc + m16);
            const float* bp1 = bp0 + 16;
#pragma unroll
            for (int h = 0; h < 16; ++h) {
                b0[h] = (_Float16)bp0[(size_t)h * 64]; // K = kk + hi*16 + h
                b1[h] = (_Float16)bp1[(size_t)h * 64];
            }
            acc00 = __builtin_amdgcn_wmma_f32_16x16x32_f16(false, a0, false, b0,
                                                           (short)0, acc00, false, false);
            acc01 = __builtin_amdgcn_wmma_f32_16x16x32_f16(false, a0, false, b1,
                                                           (short)0, acc01, false, false);
            acc10 = __builtin_amdgcn_wmma_f32_16x16x32_f16(false, a1, false, b0,
                                                           (short)0, acc10, false, false);
            acc11 = __builtin_amdgcn_wmma_f32_16x16x32_f16(false, a1, false, b1,
                                                           (short)0, acc11, false, false);
        }
        __syncthreads();            // buffer free before TDM overwrites it
    }

    const int c0 = col0 + wc + m16;
    const int c1 = c0 + 16;
    const float bv0 = bias ? bias[c0 & biasMask] : 0.0f;
    const float bv1 = bias ? bias[c1 & biasMask] : 0.0f;
#pragma unroll
    for (int r = 0; r < 8; ++r) {
        const int r0 = row0 + wr + r + 8 * hi;
        const int r1 = r0 + 16;
        float v00 = acc00[r] + bv0, v01 = acc01[r] + bv1;
        float v10 = acc10[r] + bv0, v11 = acc11[r] + bv1;
        if (relu) {
            v00 = fmaxf(v00, 0.0f); v01 = fmaxf(v01, 0.0f);
            v10 = fmaxf(v10, 0.0f); v11 = fmaxf(v11, 0.0f);
        }
        C[(size_t)r0 * N + c0] = v00;
        C[(size_t)r0 * N + c1] = v01;
        C[(size_t)r1 * N + c0] = v10;
        C[(size_t)r1 * N + c1] = v11;
    }
}

// =====================================================================
// Fused VQ: argmin_k( ||e_k||^2 - 2 p.e_k ) per pred row; never
// materializes the 8192x8192 score matrix (~268 MB HBM saved).
// =====================================================================
__global__ void vq_argmin_kernel(const float* __restrict__ pred,
                                 const float* __restrict__ embeds,
                                 const float* __restrict__ enorm,
                                 int* __restrict__ classes) {
    const int lane = threadIdx.x;
    const int m16  = lane & 15;
    const int hi   = lane >> 4;
    const int row0 = blockIdx.x * 16;

    const float* ap = pred + (size_t)(row0 + m16) * EDIM + hi * 16;
    v16h a0, a1;
#pragma unroll
    for (int i = 0; i < 8; ++i) {
        a0[i]     = (_Float16)ap[i];
        a0[8 + i] = (_Float16)ap[8 + i];
        a1[i]     = (_Float16)ap[32 + i];
        a1[8 + i] = (_Float16)ap[40 + i];
    }

    float best[8];
    int   bidx[8];
#pragma unroll
    for (int r = 0; r < 8; ++r) { best[r] = 3.0e38f; bidx[r] = 0; }

    for (int nt = 0; nt < KCODES / 16; nt += 2) {
        const int kidx0 = nt * 16 + m16;
        const int kidx1 = kidx0 + 16;
        const float* eb0 = embeds + (size_t)kidx0 * EDIM + hi * 16;
        const float* eb1 = embeds + (size_t)kidx1 * EDIM + hi * 16;
        v16h b00, b01, b10, b11;
#pragma unroll
        for (int h = 0; h < 16; ++h) {
            b00[h] = (_Float16)eb0[h];
            b01[h] = (_Float16)eb0[32 + h];
            b10[h] = (_Float16)eb1[h];
            b11[h] = (_Float16)eb1[32 + h];
        }
        v8f c0 = {}, c1 = {};
        c0 = __builtin_amdgcn_wmma_f32_16x16x32_f16(false, a0, false, b00,
                                                    (short)0, c0, false, false);
        c1 = __builtin_amdgcn_wmma_f32_16x16x32_f16(false, a0, false, b10,
                                                    (short)0, c1, false, false);
        c0 = __builtin_amdgcn_wmma_f32_16x16x32_f16(false, a1, false, b01,
                                                    (short)0, c0, false, false);
        c1 = __builtin_amdgcn_wmma_f32_16x16x32_f16(false, a1, false, b11,
                                                    (short)0, c1, false, false);
        const float en0 = enorm[kidx0];
        const float en1 = enorm[kidx1];
#pragma unroll
        for (int r = 0; r < 8; ++r) {
            const float s0 = en0 - 2.0f * c0[r];
            const float s1 = en1 - 2.0f * c1[r];
            if (s0 < best[r]) { best[r] = s0; bidx[r] = kidx0; }
            if (s1 < best[r]) { best[r] = s1; bidx[r] = kidx1; }
        }
    }

#pragma unroll
    for (int r = 0; r < 8; ++r) {
        float bv = best[r];
        int   bi = bidx[r];
        for (int off = 8; off >= 1; off >>= 1) {    // stays inside 16-lane group
            const float ov = __shfl_xor(bv, off, 32);
            const int   oi = __shfl_xor(bi, off, 32);
            if (ov < bv || (ov == bv && oi < bi)) { bv = ov; bi = oi; }
        }
        if (m16 == 0) classes[row0 + r + 8 * hi] = bi;
    }
}

// ---------------- small helper kernels ----------------

__global__ void enorm_kernel(const float* __restrict__ embeds,
                             float* __restrict__ enorm) {
    const int k = blockIdx.x * blockDim.x + threadIdx.x;
    if (k >= KCODES) return;
    const float* e = embeds + (size_t)k * EDIM;
    float s = 0.0f;
#pragma unroll 8
    for (int i = 0; i < EDIM; ++i) s += e[i] * e[i];
    enorm[k] = s;
}

// Codebook gather via async DMA: one 16B chunk per thread into its private
// LDS slot (global_load_async_to_lds_b128 / ASYNCcnt), then coalesced store.
__global__ void gather_async_kernel(const int* __restrict__ classes,
                                    const float* __restrict__ embeds,
                                    float* __restrict__ collected) {
    __shared__ char smem[256 * 16];
    const int gt = blockIdx.x * blockDim.x + threadIdx.x;   // NROWS*16 chunks
    const int r  = gt >> 4;          // pred row
    const int c  = gt & 15;          // 16B chunk within the 256B code row
    const unsigned ldsOff =
        (unsigned)(unsigned long long)(const void*)smem + threadIdx.x * 16u;
    const unsigned long long ga =
        (unsigned long long)(const void*)embeds +
        (unsigned long long)classes[r] * 256ull + (unsigned)(c * 16);
    asm volatile("global_load_async_to_lds_b128 %0, %1, off"
                 :: "v"(ldsOff), "v"(ga) : "memory");
    asm volatile("s_wait_asynccnt 0x0" ::: "memory");
    const float4 v = *(const float4*)(smem + threadIdx.x * 16u);
    *(float4*)(collected + (size_t)r * EDIM + c * 4) = v;
}

// repack HWIO (4,4,CI,CO) -> row-major (CI, 16*CO) so deconvs become GEMMs
__global__ void repack_hwio_kernel(const float* __restrict__ W,
                                   float* __restrict__ Bt, int CI, int CO) {
    const int total = 16 * CI * CO;
    const int t = blockIdx.x * blockDim.x + threadIdx.x;
    if (t >= total) return;
    const int co = t % CO;
    const int ci = (t / CO) % CI;
    const int p  = t / (CO * CI);
    Bt[(size_t)ci * (16 * CO) + p * CO + co] = W[((size_t)p * CI + ci) * CO + co];
}

// ---------------- direct conv kernels (small K, VALU) ----------------

__global__ void conv1_kernel(const float* __restrict__ x,
                             const float* __restrict__ w,
                             const float* __restrict__ b,
                             float* __restrict__ out) {
    int t = blockIdx.x * blockDim.x + threadIdx.x;
    if (t >= BATCH * 32 * 32 * 32) return;
    const int co = t & 31;
    const int ox = (t >> 5) & 31;
    const int oy = (t >> 10) & 31;
    const int bb = t >> 15;
    float s = b[co];
#pragma unroll
    for (int ky = 0; ky < 4; ++ky) {
        const int iy = oy * 2 - 1 + ky;
        if (iy < 0 || iy >= 64) continue;
#pragma unroll
        for (int kx = 0; kx < 4; ++kx) {
            const int ix = ox * 2 - 1 + kx;
            if (ix < 0 || ix >= 64) continue;
            s += x[((size_t)bb * 64 + iy) * 64 + ix] * w[(ky * 4 + kx) * 32 + co];
        }
    }
    out[t] = fmaxf(s, 0.0f);
}

__global__ void conv2_kernel(const float* __restrict__ in,
                             const float* __restrict__ w,
                             const float* __restrict__ b,
                             float* __restrict__ out) {
    int t = blockIdx.x * blockDim.x + threadIdx.x;
    if (t >= BATCH * 16 * 16 * 64) return;
    const int co = t & 63;
    const int ox = (t >> 6) & 15;
    const int oy = (t >> 10) & 15;
    const int bb = t >> 14;
    float s = b[co];
    for (int ky = 0; ky < 4; ++ky) {
        const int iy = oy * 2 - 1 + ky;
        if (iy < 0 || iy >= 32) continue;
        for (int kx = 0; kx < 4; ++kx) {
            const int ix = ox * 2 - 1 + kx;
            if (ix < 0 || ix >= 32) continue;
            const float* ip = in + (((size_t)bb * 32 + iy) * 32 + ix) * 32;
            const float* wp = w + (size_t)(ky * 4 + kx) * 32 * 64 + co;
#pragma unroll 8
            for (int ci = 0; ci < 32; ++ci) s += ip[ci] * wp[(size_t)ci * 64];
        }
    }
    out[t] = fmaxf(s, 0.0f);
}

__global__ void conv3_kernel(const float* __restrict__ in,
                             const float* __restrict__ w,
                             const float* __restrict__ b,
                             float* __restrict__ out) {
    int t = blockIdx.x * blockDim.x + threadIdx.x;
    if (t >= BATCH * 8 * 8 * 128) return;
    const int co = t & 127;
    const int ox = (t >> 7) & 7;
    const int oy = (t >> 10) & 7;
    const int bb = t >> 13;
    float s = b[co];
    for (int ky = 0; ky < 4; ++ky) {
        const int iy = oy * 2 - 1 + ky;
        if (iy < 0 || iy >= 16) continue;
        for (int kx = 0; kx < 4; ++kx) {
            const int ix = ox * 2 - 1 + kx;
            if (ix < 0 || ix >= 16) continue;
            const float* ip = in + (((size_t)bb * 16 + iy) * 16 + ix) * 64;
            const float* wp = w + (size_t)(ky * 4 + kx) * 64 * 128 + co;
#pragma unroll 8
            for (int ci = 0; ci < 64; ++ci) s += ip[ci] * wp[(size_t)ci * 128];
        }
    }
    out[t] = fmaxf(s, 0.0f);
}

// final deconv3: dt2 blocked (4096 x 2048) -> out (256,64,64,1)
__global__ void deconv3_kernel(const float* __restrict__ dt2,
                               const float* __restrict__ w3,
                               const float* __restrict__ b3,
                               float* __restrict__ out) {
    int t = blockIdx.x * blockDim.x + threadIdx.x;
    if (t >= BATCH * 64 * 64) return;
    const int OX = t & 63;
    const int OY = (t >> 6) & 63;
    const int bb = t >> 12;
    const int oy = OY >> 2, ey = OY & 3;
    const int ox = OX >> 2, ex = OX & 3;
    const int yi = oy >> 2, dy = oy & 3;
    const int xi = ox >> 2, dx = ox & 3;
    const float* ip = dt2 + ((size_t)(bb * 16 + yi * 4 + xi) * 2048) +
                      (size_t)(dy * 4 + dx) * 128;
    const float* wp = w3 + (size_t)(ey * 4 + ex) * 128;
    float s = b3[0];
#pragma unroll 8
    for (int ci = 0; ci < 128; ++ci) s += ip[ci] * wp[ci];
    out[t] = s;
}

// =====================================================================
extern "C" void kernel_launch(void* const* d_in, const int* in_sizes, int n_in,
                              void* d_out, int out_size, void* d_ws, size_t ws_size,
                              hipStream_t stream) {
    (void)in_sizes; (void)n_in; (void)out_size; (void)ws_size;
    const float* x        = (const float*)d_in[0];
    const float* enc_w1   = (const float*)d_in[1];
    const float* enc_b1   = (const float*)d_in[2];
    const float* enc_w2   = (const float*)d_in[3];
    const float* enc_b2   = (const float*)d_in[4];
    const float* enc_w3   = (const float*)d_in[5];
    const float* enc_b3   = (const float*)d_in[6];
    const float* fc_w     = (const float*)d_in[7];
    const float* fc_b     = (const float*)d_in[8];
    const float* ffc_w    = (const float*)d_in[9];
    const float* ffc_b    = (const float*)d_in[10];
    const float* embeds   = (const float*)d_in[11];
    const float* dec_fc_w = (const float*)d_in[12];
    const float* dec_fc_b = (const float*)d_in[13];
    const float* dec_w1   = (const float*)d_in[14];
    const float* dec_b1   = (const float*)d_in[15];
    const float* dec_w2   = (const float*)d_in[16];
    const float* dec_b2   = (const float*)d_in[17];
    const float* dec_w3   = (const float*)d_in[18];
    const float* dec_b3   = (const float*)d_in[19];
    float* out = (float*)d_out;

    // ---- workspace layout (floats), lifetime-based reuse ----
    float* ws = (float*)d_ws;
    float* h1        = ws;                       // 8388608  (later dt2)
    float* h2        = h1 + 8388608;             // 4194304  (later dt1)
    float* h3        = h2 + 4194304;             // 2097152
    float* fc_out    = h3 + 2097152;             // 131072
    float* pred      = fc_out + 131072;          // 524288
    float* enorm     = pred + 524288;            // 8192
    int*   classes   = (int*)(enorm + 8192);     // 8192 ints
    float* collected = (float*)(classes + 8192); // 524288
    float* dfc       = collected + 524288;       // 131072
    float* B1        = dfc + 131072;             // 2097152 (512 x 4096)
    float* B2        = B1 + 2097152;             // 524288  (256 x 2048)
    float* dt1       = h2;                       // (256 x 4096) reuse
    float* dt2       = h1;                       // (4096 x 2048) reuse

    const int TB = 256;
    auto blocks = [](long long n, int tb) { return (unsigned)((n + tb - 1) / tb); };

    repack_hwio_kernel<<<blocks(16LL * 512 * 256, TB), TB, 0, stream>>>(dec_w1, B1, 512, 256);
    repack_hwio_kernel<<<blocks(16LL * 256 * 128, TB), TB, 0, stream>>>(dec_w2, B2, 256, 128);
    enorm_kernel<<<blocks(KCODES, TB), TB, 0, stream>>>(embeds, enorm);

    conv1_kernel<<<blocks(256LL * 32 * 32 * 32, TB), TB, 0, stream>>>(x, enc_w1, enc_b1, h1);
    conv2_kernel<<<blocks(256LL * 16 * 16 * 64, TB), TB, 0, stream>>>(h1, enc_w2, enc_b2, h2);
    conv3_kernel<<<blocks(256LL * 8 * 8 * 128, TB), TB, 0, stream>>>(h2, enc_w3, enc_b3, h3);

    auto gemm = [&](const float* A, const float* Bm, const float* bias, int biasMod,
                    float* C, int M, int N, int K, int relu) {
        dim3 g(N / 64, M / 64);
        wmma_gemm_tdm_kernel<<<g, 128, 0, stream>>>(A, Bm, bias, biasMod - 1, C, M, N, K, relu);
    };

    gemm(h3, fc_w, fc_b, 512, fc_out, 256, 512, 8192, 1);
    gemm(fc_out, ffc_w, ffc_b, 2048, pred, 256, 2048, 512, 1);

    vq_argmin_kernel<<<NROWS / 16, 32, 0, stream>>>(pred, embeds, enorm, classes);
    gather_async_kernel<<<blocks((long long)NROWS * 16, TB), TB, 0, stream>>>(classes, embeds, collected);

    gemm(collected, dec_fc_w, dec_fc_b, 512, dfc, 256, 512, 2048, 1);
    gemm(dfc, B1, dec_b1, 256, dt1, 256, 4096, 512, 1);
    gemm(dt1, B2, dec_b2, 128, dt2, 4096, 2048, 256, 1);
    deconv3_kernel<<<blocks(256LL * 64 * 64, TB), TB, 0, stream>>>(dt2, dec_w3, dec_b3, out);
}
